// CrossAttn_71184787964280
// MI455X (gfx1250) — compile-verified, matrix operand order checked by reading, not run on hardware
//
#include <hip/hip_runtime.h>

// ---------------------------------------------------------------------------
// CDNA5 (gfx1250, wave32) cross-attention forward.
//   B=4, NQ=SL=D=1024, H=16, A=64.
//   d_out = [ out (B,NQ,D) f32 | attn (B,H,NQ,SL) f32 ]
//   ws    = [ Qbf 8MB | KVbf 16MB | Obuf 16MB ]
// Matmuls: v_wmma_f32_16x16x32_bf16.  Tile staging: global_load_async_to_lds.
// ---------------------------------------------------------------------------

#define B_  4
#define NQ_ 1024
#define SL_ 1024
#define D__ 1024
#define H_  16
#define A_  64
#define HA_ 1024  // H_*A_

typedef __attribute__((ext_vector_type(16))) __bf16    v16bf;
typedef __attribute__((ext_vector_type(8)))  __bf16    v8bf;
typedef __attribute__((ext_vector_type(8)))  float     v8f;
typedef __attribute__((ext_vector_type(4)))  unsigned  u32x4;
typedef __attribute__((ext_vector_type(4)))  float     f32x4;

union FragBF {
  v16bf v;
  v8bf  h[2];
  unsigned short us[16];
};

static __device__ __forceinline__ unsigned short f2bfu(float f) {
  unsigned u = __builtin_bit_cast(unsigned, f);
  u += 0x7FFFu + ((u >> 16) & 1u);   // round-to-nearest-even
  return (unsigned short)(u >> 16);
}
static __device__ __forceinline__ v8bf ld_v8bf(const unsigned short* p) {
  return __builtin_bit_cast(v8bf, *reinterpret_cast<const u32x4*>(p));
}
static __device__ __forceinline__ v8f wmma_bf16(const FragBF& a, const FragBF& b, v8f c) {
  return __builtin_amdgcn_wmma_f32_16x16x32_bf16(false, a.v, false, b.v,
                                                 (short)0, c, false, false);
}
// LDS byte address of a __shared__ pointer (generic addr low 32 bits).
static __device__ __forceinline__ unsigned lds_addr32(const void* p) {
  return (unsigned)(unsigned long long)p;
}
// Async global->LDS copy, 16B per lane at +0 and +16 (IOFFSET applies to both).
static __device__ __forceinline__ void async_cp_32B(unsigned lds, const void* g) {
  asm volatile("global_load_async_to_lds_b128 %0, %1, off\n\t"
               "global_load_async_to_lds_b128 %0, %1, off offset:16"
               :: "v"(lds), "v"(g) : "memory");
}
static __device__ __forceinline__ void async_cp_64B(unsigned lds, const void* g) {
  asm volatile("global_load_async_to_lds_b128 %0, %1, off\n\t"
               "global_load_async_to_lds_b128 %0, %1, off offset:16\n\t"
               "global_load_async_to_lds_b128 %0, %1, off offset:32\n\t"
               "global_load_async_to_lds_b128 %0, %1, off offset:48"
               :: "v"(lds), "v"(g) : "memory");
}
static __device__ __forceinline__ void wait_async0() {
  asm volatile("s_wait_asynccnt 0x0" ::: "memory");
}

// ---------------------------------------------------------------------------
// Generic GEMM: C[M,N] = A[M,K] * B[K,N], A/B f32 row-major.
// Block = 128 threads (4 waves), 64x64 tile; K chunks of 32.
// A tile staged raw-f32 via async copy (convert at fragment load);
// B tile transposed+converted to bf16 by VALU while the async copy flies.
// ---------------------------------------------------------------------------
template <bool BF16OUT>
__global__ void gemm_wmma_bf16(const float* __restrict__ A,
                               const float* __restrict__ Bm,
                               void* __restrict__ Cout,
                               int M, int N, int K) {
  __shared__ float          Af32[64][32];  // [m][k] f32 (async-staged)
  __shared__ unsigned short Bt[64][32];    // [n][k] bf16 (B transposed)
  const int t    = threadIdx.x;
  const int lane = t & 31, w = t >> 5;
  const int ln   = lane & 15, hi = lane >> 4;
  const int tm   = blockIdx.x * 64, tn = blockIdx.y * 64;

  const v8f vz = {0.f, 0.f, 0.f, 0.f, 0.f, 0.f, 0.f, 0.f};
  v8f acc[4];
#pragma unroll
  for (int i = 0; i < 4; ++i) acc[i] = vz;

  const int arow = t >> 1, ac0 = (t & 1) * 16;        // A stage: 64B/thread
  const unsigned alds = lds_addr32(&Af32[arow][ac0]);
  const int bk = t >> 2, bc0 = (t & 3) * 16;          // B stage: 16 elts/thread

  for (int k0 = 0; k0 < K; k0 += 32) {
    // A tile: issue async copy (raw f32, straight layout)
    async_cp_64B(alds, A + (size_t)(tm + arow) * K + k0 + ac0);

    // B tile: 32 k x 64 n, transpose+convert into Bt[n][k] (overlaps async)
    {
      const float* src = Bm + (size_t)(k0 + bk) * N + tn + bc0;
#pragma unroll
      for (int j = 0; j < 16; ++j) Bt[bc0 + j][bk] = f2bfu(src[j]);
    }
    wait_async0();
    __syncthreads();

    FragBF af;  // convert f32 LDS tile -> bf16 fragment
    {
      const float* ar = &Af32[w * 16 + ln][0];
      f32x4 y0 = *(const f32x4*)(ar + hi * 8);
      f32x4 y1 = *(const f32x4*)(ar + hi * 8 + 4);
      f32x4 y2 = *(const f32x4*)(ar + 16 + hi * 8);
      f32x4 y3 = *(const f32x4*)(ar + 16 + hi * 8 + 4);
#pragma unroll
      for (int j = 0; j < 4; ++j) {
        af.us[j]      = f2bfu(y0[j]);
        af.us[4 + j]  = f2bfu(y1[j]);
        af.us[8 + j]  = f2bfu(y2[j]);
        af.us[12 + j] = f2bfu(y3[j]);
      }
    }
#pragma unroll
    for (int nt = 0; nt < 4; ++nt) {
      FragBF bf;
      bf.h[0] = ld_v8bf(&Bt[nt * 16 + ln][hi * 8]);
      bf.h[1] = ld_v8bf(&Bt[nt * 16 + ln][16 + hi * 8]);
      acc[nt] = wmma_bf16(af, bf, acc[nt]);
    }
    __syncthreads();
  }

#pragma unroll
  for (int nt = 0; nt < 4; ++nt) {
#pragma unroll
    for (int r = 0; r < 8; ++r) {
      const int row = tm + w * 16 + r + hi * 8;
      const int col = tn + nt * 16 + ln;
      const float v = acc[nt][r];
      if (BF16OUT)
        ((unsigned short*)Cout)[(size_t)row * N + col] = f2bfu(v);
      else
        ((float*)Cout)[(size_t)row * N + col] = v;
    }
  }
}

// ---------------------------------------------------------------------------
// Scores + masked softmax. grid = (B*H, NQ/64); block = 128 (4 waves).
// Wave handles 16 q rows over full SL. attn region doubles as score scratch.
// ---------------------------------------------------------------------------
__global__ void attn_scores_softmax(const unsigned short* __restrict__ Qbf,
                                    const unsigned short* __restrict__ KVbf,
                                    const unsigned char* __restrict__ mask,
                                    float* __restrict__ attn) {
  const int bh = blockIdx.x;
  const int b = bh >> 4, h = bh & 15;
  const int t = threadIdx.x, lane = t & 31, w = t >> 5;
  const int ln = lane & 15, hi = lane >> 4;
  const int q0 = blockIdx.y * 64 + w * 16;
  float* at = attn + (size_t)bh * NQ_ * SL_;

  // Q fragment for this wave's 16 rows (A=64 -> two K=32 chunks), loaded once.
  FragBF af[2];
#pragma unroll
  for (int kc = 0; kc < 2; ++kc)
#pragma unroll
    for (int hh = 0; hh < 2; ++hh)
      af[kc].h[hh] = ld_v8bf(Qbf + (size_t)(b * NQ_ + q0 + ln) * HA_ +
                             h * A_ + kc * 32 + hh * 16 + hi * 8);

  float pmax[8];
#pragma unroll
  for (int r = 0; r < 8; ++r) pmax[r] = -3.0e38f;

  // Phase 1: scores = (Q K^T)/8, mask, write raw, track row max.
  for (int st = 0; st < SL_ / 16; ++st) {
    const int s0 = st * 16;
    // prefetch next tile's mask bytes (global_prefetch_b8)
    if (st + 1 < SL_ / 16)
      __builtin_prefetch(mask + (size_t)(b * NQ_ + q0 + hi * 8) * SL_ + s0 + 16 + ln, 0, 0);

    FragBF bf0, bf1;  // K rows are exactly the B^T layout: no transpose needed
#pragma unroll
    for (int hh = 0; hh < 2; ++hh) {
      const unsigned short* kr =
          KVbf + (size_t)(b * SL_ + s0 + ln) * (2 * HA_) + h * A_;
      bf0.h[hh] = ld_v8bf(kr + hh * 16 + hi * 8);
      bf1.h[hh] = ld_v8bf(kr + 32 + hh * 16 + hi * 8);
    }
    v8f acc = {0.f, 0.f, 0.f, 0.f, 0.f, 0.f, 0.f, 0.f};
    acc = wmma_bf16(af[0], bf0, acc);
    acc = wmma_bf16(af[1], bf1, acc);
#pragma unroll
    for (int r = 0; r < 8; ++r) {
      const int q = q0 + r + hi * 8;
      const int s = s0 + ln;
      float v = acc[r] * 0.125f;  // 1/sqrt(64)
      if (mask[(size_t)(b * NQ_ + q) * SL_ + s]) v = -1.0e32f;
      at[(size_t)q * SL_ + s] = v;
      pmax[r] = fmaxf(pmax[r], v);
    }
  }
  // Row max across the 16-lane N-group of the C layout.
#pragma unroll
  for (int r = 0; r < 8; ++r)
#pragma unroll
    for (int off = 1; off < 16; off <<= 1)
      pmax[r] = fmaxf(pmax[r], __shfl_xor(pmax[r], off, 32));

  // Phase 2: sum of exp (reads stay L2-resident: 256KB per block).
  float psum[8];
#pragma unroll
  for (int r = 0; r < 8; ++r) psum[r] = 0.0f;
  for (int st = 0; st < SL_ / 16; ++st) {
#pragma unroll
    for (int r = 0; r < 8; ++r) {
      const int q = q0 + r + hi * 8;
      const int s = st * 16 + ln;
      psum[r] += __expf(at[(size_t)q * SL_ + s] - pmax[r]);
    }
  }
#pragma unroll
  for (int r = 0; r < 8; ++r) {
#pragma unroll
    for (int off = 1; off < 16; off <<= 1)
      psum[r] += __shfl_xor(psum[r], off, 32);
    psum[r] = 1.0f / psum[r];
  }

  // Phase 3: write normalized attn.
  for (int st = 0; st < SL_ / 16; ++st) {
#pragma unroll
    for (int r = 0; r < 8; ++r) {
      const int q = q0 + r + hi * 8;
      const int s = st * 16 + ln;
      const size_t idx = (size_t)q * SL_ + s;
      at[idx] = __expf(at[idx] - pmax[r]) * psum[r];
    }
  }
}

// ---------------------------------------------------------------------------
// O = attn @ V per (b,h). grid = (B*H, NQ/64); block = 128 (4 waves).
// V tile async-copied row-major into LDS; transpose happens in the per-lane
// B-fragment gather. attn rows converted f32->bf16 on the fly (A fragment).
// ---------------------------------------------------------------------------
__global__ void attn_v_wmma(const float* __restrict__ attn,
                            const unsigned short* __restrict__ KVbf,
                            float* __restrict__ O) {
  __shared__ unsigned short Vrow[32][64];  // [s][a] bf16, async-staged
  const int bh = blockIdx.x;
  const int b = bh >> 4, h = bh & 15;
  const int t = threadIdx.x, lane = t & 31, w = t >> 5;
  const int ln = lane & 15, hi = lane >> 4;
  const int q0 = blockIdx.y * 64 + w * 16;
  const float* at = attn + (size_t)bh * NQ_ * SL_;

  const v8f vz = {0.f, 0.f, 0.f, 0.f, 0.f, 0.f, 0.f, 0.f};
  v8f acc[4];
#pragma unroll
  for (int i = 0; i < 4; ++i) acc[i] = vz;

  const int sc = t >> 2, c0 = (t & 3) * 16;  // V stage: 32B/thread
  const unsigned vlds = lds_addr32(&Vrow[sc][c0]);

  for (int k0 = 0; k0 < SL_; k0 += 32) {
    // stage V tile (32 s x 64 a), straight copy, async
    async_cp_32B(vlds,
                 KVbf + (size_t)(b * SL_ + k0 + sc) * (2 * HA_) + HA_ + h * A_ + c0);

    FragBF af;  // attn rows, f32 -> bf16 (overlaps the async copy)
    {
      const float* arow = at + (size_t)(q0 + ln) * SL_ + k0;
      f32x4 x0 = *(const f32x4*)(arow + hi * 8);
      f32x4 x1 = *(const f32x4*)(arow + hi * 8 + 4);
      f32x4 x2 = *(const f32x4*)(arow + 16 + hi * 8);
      f32x4 x3 = *(const f32x4*)(arow + 16 + hi * 8 + 4);
#pragma unroll
      for (int j = 0; j < 4; ++j) {
        af.us[j]      = f2bfu(x0[j]);
        af.us[4 + j]  = f2bfu(x1[j]);
        af.us[8 + j]  = f2bfu(x2[j]);
        af.us[12 + j] = f2bfu(x3[j]);
      }
    }
    wait_async0();
    __syncthreads();

#pragma unroll
    for (int nt = 0; nt < 4; ++nt) {
      FragBF bf;  // gather-transpose: element (k,n) from Vrow[k][n]
      const int n = nt * 16 + ln;
#pragma unroll
      for (int e = 0; e < 8; ++e) {
        bf.us[e]     = Vrow[hi * 8 + e][n];
        bf.us[8 + e] = Vrow[16 + hi * 8 + e][n];
      }
      acc[nt] = wmma_bf16(af, bf, acc[nt]);
    }
    __syncthreads();
  }

#pragma unroll
  for (int nt = 0; nt < 4; ++nt)
#pragma unroll
    for (int r = 0; r < 8; ++r)
      O[(size_t)(b * NQ_ + q0 + r + hi * 8) * HA_ + h * A_ + nt * 16 + ln] =
          acc[nt][r];
}

// ---------------------------------------------------------------------------
extern "C" void kernel_launch(void* const* d_in, const int* in_sizes, int n_in,
                              void* d_out, int out_size, void* d_ws, size_t ws_size,
                              hipStream_t stream) {
  (void)in_sizes; (void)n_in; (void)out_size; (void)ws_size;
  const float* iQ  = (const float*)d_in[0];
  const float* iK  = (const float*)d_in[1];
  const unsigned char* mask = (const unsigned char*)d_in[2];
  const float* Wq  = (const float*)d_in[3];
  const float* Wkv = (const float*)d_in[4];
  const float* Wo  = (const float*)d_in[5];

  float* out  = (float*)d_out;                         // (B,NQ,D)
  float* attn = out + (size_t)B_ * NQ_ * D__;          // (B,H,NQ,SL)

  char* ws = (char*)d_ws;
  unsigned short* Qbf  = (unsigned short*)ws;                        // 8 MB
  unsigned short* KVbf = (unsigned short*)(ws + (size_t)(8  << 20)); // 16 MB
  float*          Obuf = (float*)(ws + (size_t)(24 << 20));          // 16 MB

  const dim3 blk(128);
  // Q = iQ @ Wq  (4096x1024 @ 1024x1024) -> bf16
  gemm_wmma_bf16<true><<<dim3(64, 16), blk, 0, stream>>>(iQ, Wq, Qbf,
                                                         B_ * NQ_, HA_, D__);
  // KV = iK @ Wkv (4096x1024 @ 1024x2048) -> bf16 (K | V split by column)
  gemm_wmma_bf16<true><<<dim3(64, 32), blk, 0, stream>>>(iK, Wkv, KVbf,
                                                         B_ * SL_, 2 * HA_, D__);
  // scores + masked softmax -> attn (also the 2nd output)
  attn_scores_softmax<<<dim3(B_ * H_, NQ_ / 64), blk, 0, stream>>>(Qbf, KVbf,
                                                                   mask, attn);
  // O = attn @ V
  attn_v_wmma<<<dim3(B_ * H_, NQ_ / 64), blk, 0, stream>>>(attn, KVbf, Obuf);
  // out = O @ Wo (4096x1024 @ 1024x1024) -> f32
  gemm_wmma_bf16<false><<<dim3(64, 16), blk, 0, stream>>>(Obuf, Wo, out,
                                                          B_ * NQ_, D__, HA_);
}
